// GPTQLinear_74947179315668
// MI455X (gfx1250) — compile-verified, matrix operand order checked by reading, not run on hardware
//
#include <hip/hip_runtime.h>

typedef __attribute__((ext_vector_type(16))) __bf16 v16bf;
typedef __attribute__((ext_vector_type(8)))  __bf16 v8bf;
typedef __attribute__((ext_vector_type(8)))  float  v8f;

constexpr int IN_F  = 4096;
constexpr int OUT_F = 11008;
constexpr int GS    = 128;
constexpr int M_TOK = 8192;

constexpr int BLOCK_M  = 512;   // rows of x per block (16 waves x 32 rows)
constexpr int BLOCK_N  = 64;    // output features per block
constexpr int NTHREADS = 512;   // 16 waves
constexpr int LDB      = 136;   // padded k-stride (elems): 272 B, 16B-aligned, bank stride 4

// ---- pre-pass: x fp32 -> bf16 into workspace (done once; reused by all 172 N-blocks) ----
__global__ __launch_bounds__(256)
void x_to_bf16_kernel(const float* __restrict__ x, __bf16* __restrict__ xb)
{
    const long i = ((long)blockIdx.x * 256 + threadIdx.x) * 8;
    const float4 a = *(const float4*)(x + i);
    const float4 b = *(const float4*)(x + i + 4);
    v8bf t;
    t[0] = (__bf16)a.x; t[1] = (__bf16)a.y; t[2] = (__bf16)a.z; t[3] = (__bf16)a.w;
    t[4] = (__bf16)b.x; t[5] = (__bf16)b.y; t[6] = (__bf16)b.z; t[7] = (__bf16)b.w;
    *(v8bf*)(xb + i) = t;
}

// dequant 8 int4 (one packed dword) -> 8 bf16; fma folds zero-point & group scale exactly:
// val = fma(nib, s, -8s); nibble->float via (u >> 8j) & 0xFF (v_cvt_f32_ubyteN pattern)
__device__ __forceinline__ void dequant8(unsigned int w, float s, float c8, __bf16* dst)
{
    const unsigned int lo = w & 0x0F0F0F0Fu;           // even k (low nibbles)
    const unsigned int hi = (w >> 4) & 0x0F0F0F0Fu;    // odd  k (high nibbles)
    v8bf t;
#pragma unroll
    for (int j = 0; j < 4; ++j) {
        const float fl = (float)((lo >> (8 * j)) & 0xFFu);
        const float fh = (float)((hi >> (8 * j)) & 0xFFu);
        t[2 * j + 0] = (__bf16)__builtin_fmaf(fl, s, c8);
        t[2 * j + 1] = (__bf16)__builtin_fmaf(fh, s, c8);
    }
    *(v8bf*)dst = t;
}

// A fragment: 16 k-contiguous activations for one row, straight from global memory.
template <bool PRECONV>
__device__ __forceinline__ v16bf load_afrag(const void* __restrict__ xin, long elem_off)
{
    if constexpr (PRECONV) {
        return *(const v16bf*)((const __bf16*)xin + elem_off);   // 32B, 2x global_load_b128
    } else {
        const float* p = (const float*)xin + elem_off;
        v16bf t;
#pragma unroll
        for (int i = 0; i < 16; i += 4) {
            const float4 v = *(const float4*)(p + i);
            t[i + 0] = (__bf16)v.x; t[i + 1] = (__bf16)v.y;
            t[i + 2] = (__bf16)v.z; t[i + 3] = (__bf16)v.w;
        }
        return t;
    }
}

// 16 waves stacked along M (wave tile 32x64 = 2x4 frags); only dequantized B lives in LDS.
template <bool PRECONV>
__global__ __launch_bounds__(NTHREADS)
void gptq_w4a16_wmma_kernel(const void* __restrict__ xin,
                            const unsigned char* __restrict__ qw,
                            const float* __restrict__ scales,
                            const float* __restrict__ bias,
                            float* __restrict__ out)
{
    __shared__ __bf16 Bs[BLOCK_N * LDB];   // [n][k-in-group], dequantized

    const int tid  = threadIdx.x;
    const int wave = tid >> 5;             // 0..15 : M position
    const int lane = tid & 31;
    const int lg   = lane >> 4;
    const int lr   = lane & 15;

    const int m0 = blockIdx.y * BLOCK_M;
    const int n0 = blockIdx.x * BLOCK_N;

    // B loader: thread -> (row n, 8-byte part); part bp covers group-k [bp*16, bp*16+16)
    const int  bn = tid >> 3;              // 0..63
    const int  bp = tid & 7;               // 0..7
    const long gn = n0 + bn;
    const unsigned char* qrow = qw + gn * (IN_F / 2) + bp * 8;
    const float*         srow = scales + gn * (IN_F / GS);
    __bf16*              bdst = &Bs[bn * LDB + bp * 16];

    // A fragment row bases (per fm), k advances by group
    long arow[2];
#pragma unroll
    for (int fm = 0; fm < 2; ++fm)
        arow[fm] = (long)(m0 + wave * 32 + fm * 16 + lr) * IN_F;

    v8f acc[2][4] = {};

    constexpr int NGROUP = IN_F / GS;      // 32
    for (int g = 0; g < NGROUP; ++g) {
        // ---- global loads (latency overlaps first barrier wait) ----
        const uint2 pkw = *(const uint2*)(qrow + g * (GS / 2));
        const float s   = srow[g];
        const float c8  = -8.0f * s;

        if (g + 1 < NGROUP)                 // global_prefetch_b8 next weight tile
            __builtin_prefetch(qrow + (g + 1) * (GS / 2), 0, 0);

        __syncthreads();                    // all waves done reading previous B group

        dequant8(pkw.x, s, c8, bdst);       // 16 int4 -> 16 bf16 per thread
        dequant8(pkw.y, s, c8, bdst + 8);

        __syncthreads();

        // ---- 4 k-steps x 8 WMMA; A straight from global, B from LDS ----
        const long koffA = (long)g * GS + lg * 16;
#pragma unroll
        for (int ss = 0; ss < 4; ++ss) {
            const int ko = ss * 32 + lg * 16;
            v16bf af[2], bfv[4];
#pragma unroll
            for (int fm = 0; fm < 2; ++fm)
                af[fm] = load_afrag<PRECONV>(xin, arow[fm] + koffA + ss * 32);
#pragma unroll
            for (int fn = 0; fn < 4; ++fn)
                bfv[fn] = *(const v16bf*)&Bs[(fn * 16 + lr) * LDB + ko];
#pragma unroll
            for (int fm = 0; fm < 2; ++fm)
#pragma unroll
                for (int fn = 0; fn < 4; ++fn)
                    acc[fm][fn] = __builtin_amdgcn_wmma_f32_16x16x32_bf16(
                        false, af[fm], false, bfv[fn],
                        (short)0, acc[fm][fn], false, false);
        }
    }

    // ---- epilogue: bias + store (C/D layout: m = reg + 8*lg, n = lr) ----
#pragma unroll
    for (int fn = 0; fn < 4; ++fn) {
        const int   gnn = n0 + fn * 16 + lr;
        const float bv  = bias[gnn];
#pragma unroll
        for (int fm = 0; fm < 2; ++fm) {
#pragma unroll
            for (int r = 0; r < 8; ++r) {
                const int gm = m0 + wave * 32 + fm * 16 + lg * 8 + r;
                out[(long)gm * OUT_F + gnn] = acc[fm][fn][r] + bv;
            }
        }
    }
}

extern "C" void kernel_launch(void* const* d_in, const int* in_sizes, int n_in,
                              void* d_out, int out_size, void* d_ws, size_t ws_size,
                              hipStream_t stream)
{
    (void)in_sizes; (void)n_in; (void)out_size;
    const float*         x      = (const float*)d_in[0];
    const unsigned char* qw     = (const unsigned char*)d_in[1];
    const float*         scales = (const float*)d_in[2];
    const float*         bias   = (const float*)d_in[3];
    float*               out    = (float*)d_out;

    dim3 grid(OUT_F / BLOCK_N, M_TOK / BLOCK_M);   // 172 x 16 (N fastest -> L2 reuse of x slab)
    dim3 block(NTHREADS);

    const size_t xb_bytes = (size_t)M_TOK * IN_F * 2;   // 67 MB bf16 copy of x
    if (ws_size >= xb_bytes) {
        const long nconv = ((long)M_TOK * IN_F) / (256 * 8);
        hipLaunchKernelGGL(x_to_bf16_kernel, dim3((unsigned)nconv), dim3(256), 0, stream,
                           x, (__bf16*)d_ws);
        hipLaunchKernelGGL((gptq_w4a16_wmma_kernel<true>), grid, block, 0, stream,
                           (const void*)d_ws, qw, scales, bias, out);
    } else {
        hipLaunchKernelGGL((gptq_w4a16_wmma_kernel<false>), grid, block, 0, stream,
                           (const void*)x, qw, scales, bias, out);
    }
}
